// HaloAttention_30554397343790
// MI455X (gfx1250) — matrix operand order repaired and checked by reading, not compiled, not run on hardware
//
#include <hip/hip_runtime.h>
#include <hip/hip_bf16.h>
#include <math.h>

// ---------------------------------------------------------------------------
// HaloAttention for MI455X (gfx1250): all matmuls on v_wmma_f32_16x16x32_f16.
// Round 2: batched tile loads (MLP before s_wait), vectorized b128 data paths.
// ---------------------------------------------------------------------------

typedef __attribute__((ext_vector_type(16))) _Float16 v16h;
typedef __attribute__((ext_vector_type(8)))  float    v8f;
typedef _Float16 half_t;

#define BATCH   8
#define HH      128
#define WW      128
#define CC      256
#define N_HEADS 8
#define QK_CH   256
#define KV_CHN  512
#define M_TOT   (BATCH * HH * WW)   // 131072

union Frag { v16h v; unsigned int u[8]; };
union H32  { uint4 q4[4]; half_t h[32]; };
union F4   { float4 v; float f[4]; };

// A-matrix fragment (16-bit, 16x32): lane holds row M = lane%16.
// VGPR j halves (2j,2j+1) map to K = (lane>=16)*16 + j*2 (+1).
__device__ __forceinline__ v16h frag_a(const half_t* row, int hi) {
  const unsigned int* r = (const unsigned int*)row;
  Frag f;
#pragma unroll
  for (int j = 0; j < 8; ++j)
    f.u[j] = r[(hi << 3) + j];
  return f.v;
}

// B-matrix fragment (16-bit, 32x16): lane holds column N = lane%16,
// halves hold K = (lane>=16)*16 + halfIdx. Reads [N][K] (K contiguous).
__device__ __forceinline__ v16h frag_b(const half_t* col, int hi) {
  const unsigned int* r = (const unsigned int*)col;
  Frag f;
#pragma unroll
  for (int j = 0; j < 8; ++j)
    f.u[j] = r[(hi << 3) + j];
  return f.v;
}

__device__ __forceinline__ v8f vzero8() {
  v8f z = {0.f, 0.f, 0.f, 0.f, 0.f, 0.f, 0.f, 0.f};
  return z;
}

// ---------------------------------------------------------------------------
// 1) Continuous-position-bias table: tab[head][q(64)][k(64)]
// ---------------------------------------------------------------------------
__global__ __launch_bounds__(256) void k_rel_bias(const float* __restrict__ w1,
                                                  const float* __restrict__ b1,
                                                  const float* __restrict__ w2,
                                                  float* __restrict__ tab) {
  int idx = blockIdx.x * 256 + threadIdx.x;   // 4096 = 64*64
  int kk = idx & 63, qq = idx >> 6;
  int qy = qq >> 3, qx = qq & 7, ky = kk >> 3, kx = kk & 7;
  float ry = (float)qy - (((float)ky - 2.0f) * 2.0f + 0.5f);
  float rx = (float)qx - (((float)kx - 2.0f) * 2.0f + 0.5f);
  auto tr = [](float r) {
    r = r * (8.0f / 7.0f);
    float a = log2f(fabsf(r) + 1.0f) * (1.0f / log2f(8.0f));
    return r > 0.f ? a : (r < 0.f ? -a : 0.0f);
  };
  float r0 = tr(ry), r1 = tr(rx);
  float o[N_HEADS] = {0.f, 0.f, 0.f, 0.f, 0.f, 0.f, 0.f, 0.f};
  for (int j = 0; j < 512; ++j) {
    float h = fmaxf(r0 * w1[j] + r1 * w1[512 + j] + b1[j], 0.0f);
#pragma unroll
    for (int t = 0; t < N_HEADS; ++t) o[t] += h * w2[j * N_HEADS + t];
  }
#pragma unroll
  for (int t = 0; t < N_HEADS; ++t)
    tab[((size_t)t * 64 + qq) * 64 + kk] = 16.0f / (1.0f + __expf(-o[t]));
}

// ---------------------------------------------------------------------------
// 2) QKV GEMM:  [131072 x 256] @ [256 x 768] -> q(f16,+bias) / kv(f16)
// ---------------------------------------------------------------------------
__global__ __launch_bounds__(256) void k_qkv_gemm(const float* __restrict__ X,
                                                  const float* __restrict__ Wq,
                                                  const float* __restrict__ q_bias,
                                                  half_t* __restrict__ q_out,
                                                  half_t* __restrict__ kv_out) {
  const int KD = 256, ND = 768, TK = 32;
  __shared__ half_t As[2][128][34];   // [m][k], padded row (17 dwords)
  __shared__ half_t Bs[2][128][34];   // [n][k] (transposed), padded
  const int row0 = blockIdx.y * 128, col0 = blockIdx.x * 128;
  const int tid = threadIdx.x, lane = tid & 31, wave = tid >> 5;
  const int wr = wave >> 1, wc = wave & 1, hi = lane >> 4, ln = lane & 15;

  v8f acc[2][4];
#pragma unroll
  for (int mt = 0; mt < 2; ++mt)
#pragma unroll
    for (int nt = 0; nt < 4; ++nt) acc[mt][nt] = vzero8();

  // Issue ALL 8 global b128 loads before any wait/convert/LDS-store so the
  // memory pipe has 8 outstanding requests (s_wait_loadcnt counts down).
  auto loadt = [&](int kkk, int buf) {
    float4 fa[4], fb[4];
#pragma unroll
    for (int it = 0; it < 4; ++it) {            // A: 128x32 f32
      int idx = tid + it * 256;
      fa[it] = ((const float4*)(X + (size_t)(row0 + (idx >> 3)) * KD + kkk))[idx & 7];
    }
#pragma unroll
    for (int it = 0; it < 4; ++it) {            // B: 32x128 f32
      int idx = tid + it * 256;
      fb[it] = ((const float4*)(Wq + (size_t)(kkk + (idx >> 5)) * ND + col0))[idx & 31];
    }
#pragma unroll
    for (int it = 0; it < 4; ++it) {
      int idx = tid + it * 256;
      half_t* dst = &As[buf][idx >> 3][(idx & 7) * 4];
      dst[0] = (half_t)fa[it].x; dst[1] = (half_t)fa[it].y;
      dst[2] = (half_t)fa[it].z; dst[3] = (half_t)fa[it].w;
    }
#pragma unroll
    for (int it = 0; it < 4; ++it) {
      int idx = tid + it * 256;
      int r = idx >> 5, c4 = idx & 31;
      Bs[buf][c4 * 4 + 0][r] = (half_t)fb[it].x;
      Bs[buf][c4 * 4 + 1][r] = (half_t)fb[it].y;
      Bs[buf][c4 * 4 + 2][r] = (half_t)fb[it].z;
      Bs[buf][c4 * 4 + 3][r] = (half_t)fb[it].w;
    }
  };

  const int nk = KD / TK;
  loadt(0, 0);
  __syncthreads();
  for (int ks = 0; ks < nk; ++ks) {
    int buf = ks & 1;
    if (ks + 1 < nk) loadt((ks + 1) * TK, buf ^ 1);
    if (ks + 2 < nk) {                           // -> global_prefetch
      __builtin_prefetch(X + (size_t)(row0 + (tid & 127)) * KD + (ks + 2) * TK, 0, 1);
      __builtin_prefetch(Wq + (size_t)((ks + 2) * TK + (tid & 31)) * ND + col0, 0, 1);
    }
    v16h a[2], b[4];
#pragma unroll
    for (int mt = 0; mt < 2; ++mt)
      a[mt] = frag_a(&As[buf][wr * 32 + mt * 16 + ln][0], hi);
#pragma unroll
    for (int nt = 0; nt < 4; ++nt)
      b[nt] = frag_b(&Bs[buf][wc * 64 + nt * 16 + ln][0], hi);
#pragma unroll
    for (int mt = 0; mt < 2; ++mt)
#pragma unroll
      for (int nt = 0; nt < 4; ++nt)
        acc[mt][nt] = __builtin_amdgcn_wmma_f32_16x16x32_f16(
            false, a[mt], false, b[nt], (short)0, acc[mt][nt], false, false);
    __syncthreads();
  }

#pragma unroll
  for (int mt = 0; mt < 2; ++mt)
#pragma unroll
    for (int nt = 0; nt < 4; ++nt) {
      int n = col0 + wc * 64 + nt * 16 + ln;
#pragma unroll
      for (int r = 0; r < 8; ++r) {
        int m = row0 + wr * 32 + mt * 16 + hi * 8 + r;
        float v = acc[mt][nt][r];
        if (n < QK_CH) q_out[(size_t)m * QK_CH + n] = (half_t)(v + q_bias[n]);
        else           kv_out[(size_t)m * KV_CHN + (n - QK_CH)] = (half_t)v;
      }
    }
}

// ---------------------------------------------------------------------------
// 3) q <- l2n(q) * exp(min(scale, ln 100)), per (pixel, head) -- b128 paths
// ---------------------------------------------------------------------------
__global__ __launch_bounds__(256) void k_q_norm(half_t* __restrict__ q,
                                                const float* __restrict__ scale) {
  size_t i = (size_t)blockIdx.x * 256 + threadIdx.x;  // 1,048,576
  half_t* p = q + i * 32;
  H32 u;
  const uint4* s4 = (const uint4*)p;
#pragma unroll
  for (int t = 0; t < 4; ++t) u.q4[t] = s4[t];
  float v[32], s = 0.f;
#pragma unroll
  for (int j = 0; j < 32; ++j) { v[j] = (float)u.h[j]; s += v[j] * v[j]; }
  float rs = rsqrtf(fmaxf(s, 1.55e-5f));
  float ls = __expf(fminf(scale[i & 7], 4.6051702f));
#pragma unroll
  for (int j = 0; j < 32; ++j) u.h[j] = (half_t)(v[j] * rs * ls);
  uint4* d4 = (uint4*)p;
#pragma unroll
  for (int t = 0; t < 4; ++t) d4[t] = u.q4[t];
}

// ---------------------------------------------------------------------------
// 4) depthwise 3x3 stride-2 conv + BN + (k: l2n | v: +v_bias) -> kv_post f16
//    one thread per (b, oy, ox, 32-channel head group); vectorized taps.
// ---------------------------------------------------------------------------
__global__ __launch_bounds__(256) void k_dwconv(const half_t* __restrict__ kvp,
                                                const float* __restrict__ w,
                                                const float* __restrict__ gamma,
                                                const float* __restrict__ beta,
                                                const float* __restrict__ mean,
                                                const float* __restrict__ var,
                                                const float* __restrict__ v_bias,
                                                half_t* __restrict__ kvo) {
  int idx = blockIdx.x * 256 + threadIdx.x;     // 8*64*64*16
  int g = idx & 15, pos = idx >> 4;
  int ox = pos & 63, oy = (pos >> 6) & 63, b = pos >> 12;
  int c0 = g * 32;
  float acc[32];
#pragma unroll
  for (int j = 0; j < 32; ++j) acc[j] = 0.f;
  // jax 'SAME', stride 2, in=128, k=3 -> pad_lo=0, pad_hi=1: in = 2*o + k
#pragma unroll
  for (int ky = 0; ky < 3; ++ky) {
    int iy = oy * 2 + ky;
    if (iy >= HH) continue;
#pragma unroll
    for (int kx = 0; kx < 3; ++kx) {
      int ix = ox * 2 + kx;
      if (ix >= WW) continue;
      const uint4* s4 =
          (const uint4*)(kvp + (((size_t)(b * HH + iy) * WW + ix) * KV_CHN + c0));
      H32 u;
#pragma unroll
      for (int t = 0; t < 4; ++t) u.q4[t] = s4[t];
      const float4* w4 = (const float4*)(w + (size_t)(ky * 3 + kx) * KV_CHN + c0);
#pragma unroll
      for (int t = 0; t < 8; ++t) {
        F4 wv; wv.v = w4[t];
#pragma unroll
        for (int e = 0; e < 4; ++e)
          acc[t * 4 + e] += (float)u.h[t * 4 + e] * wv.f[e];
      }
    }
  }
  float out[32], ss = 0.f;
#pragma unroll
  for (int t = 0; t < 8; ++t) {
    F4 m4, v4, g4, b4;
    m4.v = ((const float4*)(mean  + c0))[t];
    v4.v = ((const float4*)(var   + c0))[t];
    g4.v = ((const float4*)(gamma + c0))[t];
    b4.v = ((const float4*)(beta  + c0))[t];
#pragma unroll
    for (int e = 0; e < 4; ++e) {
      int j = t * 4 + e;
      float x = (acc[j] - m4.f[e]) * rsqrtf(v4.f[e] + 1e-3f) * g4.f[e] + b4.f[e];
      if (c0 < QK_CH) ss += x * x;          // k half: accumulate for l2n
      out[j] = x;
    }
  }
  if (c0 < QK_CH) {
    float rs = rsqrtf(fmaxf(ss, 1.55e-5f));
#pragma unroll
    for (int j = 0; j < 32; ++j) out[j] *= rs;
  } else {
#pragma unroll
    for (int t = 0; t < 8; ++t) {
      F4 vb; vb.v = ((const float4*)(v_bias + (c0 - QK_CH)))[t];
#pragma unroll
      for (int e = 0; e < 4; ++e) out[t * 4 + e] += vb.f[e];
    }
  }
  H32 o;
#pragma unroll
  for (int j = 0; j < 32; ++j) o.h[j] = (half_t)out[j];
  uint4* d4 = (uint4*)(kvo + ((size_t)pos * KV_CHN + c0));
#pragma unroll
  for (int t = 0; t < 4; ++t) d4[t] = o.q4[t];
}

// ---------------------------------------------------------------------------
// 5) halo attention: one wave per (b, window, head).
//    S = Q·K^T (16 WMMA), +bias +mask, softmax, O = P·V (16 WMMA).
// ---------------------------------------------------------------------------
__global__ __launch_bounds__(32) void k_halo_attn(const half_t* __restrict__ qn,
                                                  const half_t* __restrict__ kvp,
                                                  const float* __restrict__ btab,
                                                  half_t* __restrict__ outp) {
  __shared__ half_t Pl[64][66];  // softmax probs [q][k], padded
  __shared__ half_t Vt[32][72];  // V transposed  [d][k-pos], padded
  int task = blockIdx.x;
  int head = task & 7, win = (task >> 3) & 255, b = task >> 11;
  int wy = win >> 4, wx = win & 15;
  int lane = threadIdx.x, hi = lane >> 4, ln = lane & 15;

  // V -> LDS transposed: batched b128 loads (OOB halo positions = 0)
  uint4 vd[2][4];
#pragma unroll
  for (int pp = 0; pp < 2; ++pp) {
    int p = lane + pp * 32;
    int gy = wy * 4 - 2 + (p >> 3), gx = wx * 4 - 2 + (p & 7);
    bool ok = ((unsigned)gy < 64u) && ((unsigned)gx < 64u);
    int cy = ok ? gy : 0, cx = ok ? gx : 0;
    const uint4* src =
        (const uint4*)(kvp + ((((size_t)b * 64 + cy) * 64 + cx) * KV_CHN + QK_CH + head * 32));
    uint4 z = {0u, 0u, 0u, 0u};
#pragma unroll
    for (int t = 0; t < 4; ++t) { uint4 d = src[t]; vd[pp][t] = ok ? d : z; }
  }
#pragma unroll
  for (int pp = 0; pp < 2; ++pp) {
    int p = lane + pp * 32;
    H32 u;
#pragma unroll
    for (int t = 0; t < 4; ++t) u.q4[t] = vd[pp][t];
#pragma unroll
    for (int d = 0; d < 32; ++d) Vt[d][p] = u.h[d];
  }

  // pad mask per (lane, nt)
  float maskv[4];
#pragma unroll
  for (int nt = 0; nt < 4; ++nt) {
    int n = nt * 16 + ln;
    int gy = wy * 4 - 2 + (n >> 3), gx = wx * 4 - 2 + (n & 7);
    maskv[nt] = (((unsigned)gy < 64u) && ((unsigned)gx < 64u)) ? 0.0f : -100.0f;
  }

  // S = Q·K^T : fragments gathered directly from global (K-pairs contiguous)
  v8f acc[4][4];
#pragma unroll
  for (int mt = 0; mt < 4; ++mt)
#pragma unroll
    for (int nt = 0; nt < 4; ++nt) acc[mt][nt] = vzero8();

  v16h af[4], bf[4];
#pragma unroll
  for (int mt = 0; mt < 4; ++mt) {
    int m = mt * 16 + ln;
    int y = wy * 8 + (m >> 3), x = wx * 8 + (m & 7);
    af[mt] = frag_a(qn + (((size_t)(b * HH + y) * WW + x) * QK_CH + head * 32), hi);
  }
#pragma unroll
  for (int nt = 0; nt < 4; ++nt) {
    int n = nt * 16 + ln;
    int gy = wy * 4 - 2 + (n >> 3), gx = wx * 4 - 2 + (n & 7);
    bool ok = ((unsigned)gy < 64u) && ((unsigned)gx < 64u);
    int cy = ok ? gy : 0, cx = ok ? gx : 0;
    const unsigned int* cu =
        (const unsigned int*)(kvp + ((((size_t)b * 64 + cy) * 64 + cx) * KV_CHN + head * 32));
    Frag f;
#pragma unroll
    for (int j = 0; j < 8; ++j) f.u[j] = ok ? cu[(hi << 3) + j] : 0u;  // cndmask
    bf[nt] = f.v;
  }
#pragma unroll
  for (int mt = 0; mt < 4; ++mt)
#pragma unroll
    for (int nt = 0; nt < 4; ++nt)
      acc[mt][nt] = __builtin_amdgcn_wmma_f32_16x16x32_f16(
          false, af[mt], false, bf[nt], (short)0, acc[mt][nt], false, false);

  // bias + mask + row softmax on f32 accumulators; store probs f16 to LDS.
#pragma unroll
  for (int mt = 0; mt < 4; ++mt) {
#pragma unroll
    for (int r = 0; r < 8; ++r) {
      int m = mt * 16 + hi * 8 + r;
      float xv[4];
#pragma unroll
      for (int nt = 0; nt < 4; ++nt) {
        int n = nt * 16 + ln;
        xv[nt] = acc[mt][nt][r] + btab[((size_t)head * 64 + m) * 64 + n] + maskv[nt];
      }
      float mx = fmaxf(fmaxf(xv[0], xv[1]), fmaxf(xv[2], xv[3]));
#pragma unroll
      for (int off = 1; off < 16; off <<= 1) mx = fmaxf(mx, __shfl_xor(mx, off, 32));
      float s = 0.f;
#pragma unroll
      for (int nt = 0; nt < 4; ++nt) { xv[nt] = __expf(xv[nt] - mx); s += xv[nt]; }
#pragma unroll
      for (int off = 1; off < 16; off <<= 1) s += __shfl_xor(s, off, 32);
      float inv = 1.0f / s;
#pragma unroll
      for (int nt = 0; nt < 4; ++nt) Pl[m][nt * 16 + ln] = (half_t)(xv[nt] * inv);
    }
  }
  __syncthreads();

  // O = P·V : K = 64 -> two 32-wide WMMA steps
  v8f acc2[4][2];
#pragma unroll
  for (int mt = 0; mt < 4; ++mt)
#pragma unroll
    for (int nt = 0; nt < 2; ++nt) acc2[mt][nt] = vzero8();
#pragma unroll
  for (int s = 0; s < 2; ++s) {
    v16h pa[4], vb[2];
#pragma unroll
    for (int mt = 0; mt < 4; ++mt) pa[mt] = frag_a(&Pl[mt * 16 + ln][s * 32], hi);
#pragma unroll
    for (int nt = 0; nt < 2; ++nt) vb[nt] = frag_b(&Vt[nt * 16 + ln][s * 32], hi);
#pragma unroll
    for (int mt = 0; mt < 4; ++mt)
#pragma unroll
      for (int nt = 0; nt < 2; ++nt)
        acc2[mt][nt] = __builtin_amdgcn_wmma_f32_16x16x32_f16(
            false, pa[mt], false, vb[nt], (short)0, acc2[mt][nt], false, false);
  }

  // scatter back to (b, y, x, head*32+d) layout (the reference transpose)
#pragma unroll
  for (int mt = 0; mt < 4; ++mt)
#pragma unroll
    for (int nt = 0; nt < 2; ++nt)
#pragma unroll
      for (int r = 0; r < 8; ++r) {
        int m = mt * 16 + hi * 8 + r;
        int d = nt * 16 + ln;
        int y = wy * 8 + (m >> 3), x = wx * 8 + (m & 7);
        outp[((size_t)(b * HH + y) * WW + x) * CC + head * 32 + d] = (half_t)acc2[mt][nt][r];
      }
}

// ---------------------------------------------------------------------------
// 6) output projection: [131072 x 256](f16) @ [256 x 256] + bias -> f32 out
// ---------------------------------------------------------------------------
__global__ __launch_bounds__(256) void k_proj_gemm(const half_t* __restrict__ A,
                                                   const float* __restrict__ Wp,
                                                   const float* __restrict__ bias,
                                                   float* __restrict__ outp) {
  const int KD = 256, ND = 256, TK = 32;
  __shared__ half_t As[2][128][34];
  __shared__ half_t Bs[2][128][34];
  const int row0 = blockIdx.y * 128, col0 = blockIdx.x * 128;
  const int tid = threadIdx.x, lane = tid & 31, wave = tid >> 5;
  const int wr = wave >> 1, wc = wave & 1, hi = lane >> 4, ln = lane & 15;

  v8f acc[2][4];
#pragma unroll
  for (int mt = 0; mt < 2; ++mt)
#pragma unroll
    for (int nt = 0; nt < 4; ++nt) acc[mt][nt] = vzero8();

  // all global loads first, then convert/store (max outstanding loads)
  auto loadt = [&](int kkk, int buf) {
    uint4 ua[2]; float4 fb[4];
#pragma unroll
    for (int it = 0; it < 2; ++it) {            // A f16: 128x32 via b128
      int idx = tid + it * 256;
      ua[it] = ((const uint4*)(A + (size_t)(row0 + (idx >> 2)) * KD + kkk))[idx & 3];
    }
#pragma unroll
    for (int it = 0; it < 4; ++it) {            // B f32: 32x128
      int idx = tid + it * 256;
      fb[it] = ((const float4*)(Wp + (size_t)(kkk + (idx >> 5)) * ND + col0))[idx & 31];
    }
#pragma unroll
    for (int it = 0; it < 2; ++it) {
      int idx = tid + it * 256;
      int r = idx >> 2, c8 = idx & 3;
      unsigned int* dst = (unsigned int*)&As[buf][r][0];
      dst[c8 * 4 + 0] = ua[it].x; dst[c8 * 4 + 1] = ua[it].y;
      dst[c8 * 4 + 2] = ua[it].z; dst[c8 * 4 + 3] = ua[it].w;
    }
#pragma unroll
    for (int it = 0; it < 4; ++it) {
      int idx = tid + it * 256;
      int r = idx >> 5, c4 = idx & 31;
      Bs[buf][c4 * 4 + 0][r] = (half_t)fb[it].x;
      Bs[buf][c4 * 4 + 1][r] = (half_t)fb[it].y;
      Bs[buf][c4 * 4 + 2][r] = (half_t)fb[it].z;
      Bs[buf][c4 * 4 + 3][r] = (half_t)fb[it].w;
    }
  };

  const int nk = KD / TK;
  loadt(0, 0);
  __syncthreads();
  for (int ks = 0; ks < nk; ++ks) {
    int buf = ks & 1;
    if (ks + 1 < nk) loadt((ks + 1) * TK, buf ^ 1);
    v16h a[2], b[4];
#pragma unroll
    for (int mt = 0; mt < 2; ++mt)
      a[mt] = frag_a(&As[buf][wr * 32 + mt * 16 + ln][0], hi);
#pragma unroll
    for (int nt = 0; nt < 4; ++nt)
      b[nt] = frag_b(&Bs[buf][wc * 64 + nt * 16 + ln][0], hi);
#pragma unroll
    for (int mt = 0; mt < 2; ++mt)
#pragma unroll
      for (int nt = 0; nt < 4; ++nt)
        acc[mt][nt] = __builtin_amdgcn_wmma_f32_16x16x32_f16(
            false, a[mt], false, b[nt], (short)0, acc[mt][nt], false, false);
    __syncthreads();
  }

#pragma unroll
  for (int mt = 0; mt < 2; ++mt)
#pragma unroll
    for (int nt = 0; nt < 4; ++nt) {
      int n = col0 + wc * 64 + nt * 16 + ln;
#pragma unroll
      for (int r = 0; r < 8; ++r) {
        int m = row0 + wr * 32 + mt * 16 + hi * 8 + r;
        outp[(size_t)m * ND + n] = acc[mt][nt][r] + bias[n];
      }
    }
}

// ---------------------------------------------------------------------------
extern "C" void kernel_launch(void* const* d_in, const int* in_sizes, int n_in,
                              void* d_out, int out_size, void* d_ws, size_t ws_size,
                              hipStream_t stream) {
  (void)in_sizes; (void)n_in; (void)out_size; (void)ws_size;
  const float* inputs = (const float*)d_in[0];
  const float* qkv_w  = (const float*)d_in[1];
  const float* q_bias = (const float*)d_in[2];
  const float* v_bias = (const float*)d_in[3];
  const float* dw_w   = (const float*)d_in[4];
  const float* dw_g   = (const float*)d_in[5];
  const float* dw_b   = (const float*)d_in[6];
  const float* dw_m   = (const float*)d_in[7];
  const float* dw_v   = (const float*)d_in[8];
  const float* scale  = (const float*)d_in[9];
  const float* cpb_w1 = (const float*)d_in[10];
  const float* cpb_b1 = (const float*)d_in[11];
  const float* cpb_w2 = (const float*)d_in[12];
  const float* proj_w = (const float*)d_in[13];
  const float* proj_b = (const float*)d_in[14];

  char* ws = (char*)d_ws;
  size_t o = 0;
  half_t* q_n     = (half_t*)(ws + o); o += (size_t)M_TOT * QK_CH * 2;            // 64 MB
  half_t* kv_pre  = (half_t*)(ws + o); o += (size_t)M_TOT * KV_CHN * 2;           // 128 MB
  half_t* kv_post = (half_t*)(ws + o); o += (size_t)BATCH * 64 * 64 * KV_CHN * 2; // 32 MB
  half_t* a_out   = (half_t*)(ws + o); o += (size_t)M_TOT * CC * 2;               // 64 MB
  float*  btab    = (float*)(ws + o);  o += (size_t)N_HEADS * 64 * 64 * 4;        // 128 KB

  k_rel_bias <<<dim3(16),      dim3(256), 0, stream>>>(cpb_w1, cpb_b1, cpb_w2, btab);
  k_qkv_gemm <<<dim3(6, 1024), dim3(256), 0, stream>>>(inputs, qkv_w, q_bias, q_n, kv_pre);
  k_q_norm   <<<dim3(4096),    dim3(256), 0, stream>>>(q_n, scale);
  k_dwconv   <<<dim3(2048),    dim3(256), 0, stream>>>(kv_pre, dw_w, dw_g, dw_b, dw_m, dw_v,
                                                       v_bias, kv_post);
  k_halo_attn<<<dim3(16384),   dim3(32),  0, stream>>>(q_n, kv_post, btab, a_out);
  k_proj_gemm<<<dim3(2, 1024), dim3(256), 0, stream>>>(a_out, proj_w, proj_b, (float*)d_out);
}
